// SelfAttention_73323681677490
// MI455X (gfx1250) — compile-verified, matrix operand order checked by reading, not run on hardware
//
#include <hip/hip_runtime.h>

typedef __attribute__((ext_vector_type(16))) _Float16 v16h;
typedef __attribute__((ext_vector_type(8)))  float    v8f;

#define HEADS   8
#define HDIM    64
#define NBATCH  2
#define SEQ     4096
#define NROWS   (NBATCH * SEQ)          // 8192 total (n,s) rows
#define SM_SCALE 0.04419417382415922f   // 1/sqrt(512)

// ---------------------------------------------------------------------------
// CDNA5 WMMA helper: D(16x16,f32) = A(16x32,f16) * B(32x16,f16) + C
// ---------------------------------------------------------------------------
__device__ __forceinline__ v8f wmma_f16(v16h a, v16h b, v8f c) {
  return __builtin_amdgcn_wmma_f32_16x16x32_f16(
      /*neg_a=*/false, a, /*neg_b=*/false, b,
      /*c_mod=*/(short)0, c, /*reuse_a=*/false, /*reuse_b=*/false);
}

// A-fragment (and mirrored B-fragment) of a row-major 16x32 f16 tile:
// lane (m = lane&15, half = lane>>4) holds row m, elements 0..7 at
// k = kBase + half*8 .. +7, elements 8..15 at k = kBase + 16 + half*8 .. +7.
// Each segment is 16 contiguous bytes -> two b128 loads per lane.
__device__ __forceinline__ v16h ldfrag(const _Float16* rowPtr, int kBase, int half) {
  union { v16h v; uint4 q[2]; } u;
  u.q[0] = *(const uint4*)(rowPtr + kBase + half * 8);
  u.q[1] = *(const uint4*)(rowPtr + kBase + 16 + half * 8);
  return u.v;
}

// Same fragment but sourced from f32 memory with convert-to-f16.
__device__ __forceinline__ v16h ldfrag_f32(const float* rowPtr, int kBase, int half) {
  union { v16h v; _Float16 h[16]; } u;
  const float* p0 = rowPtr + kBase + half * 8;
  const float* p1 = rowPtr + kBase + 16 + half * 8;
#pragma unroll
  for (int i = 0; i < 8; ++i) {
    u.h[i]     = (_Float16)p0[i];
    u.h[8 + i] = (_Float16)p1[i];
  }
  return u.v;
}

// ---------------------------------------------------------------------------
// Kernel 0: Wfc (f32 512x512) -> f16
// ---------------------------------------------------------------------------
__global__ __launch_bounds__(256) void cvt_kernel(const float* __restrict__ w,
                                                  _Float16* __restrict__ wh) {
  int i = blockIdx.x * 256 + threadIdx.x;
  wh[i] = (_Float16)w[i];
}

// ---------------------------------------------------------------------------
// Kernel 1: fused QKV projection via WMMA.
//   Q,K out: f16 [N,H,S,64] row-major.  V out: f16 [N,H,64,S] (transposed).
// One wave handles one (tensor, head, 16-row block): Y(16x64) = X(16x64) @ W^T.
// ---------------------------------------------------------------------------
__global__ __launch_bounds__(256) void proj_kernel(
    const float* __restrict__ xq, const float* __restrict__ xk,
    const float* __restrict__ xv,
    const float* __restrict__ Wq, const float* __restrict__ Wk,
    const float* __restrict__ Wv,
    _Float16* __restrict__ qh, _Float16* __restrict__ kh,
    _Float16* __restrict__ vt) {
  int wave = blockIdx.x * 8 + (threadIdx.x >> 5);   // 0 .. 12287
  int lane = threadIdx.x & 31;
  int m = lane & 15, half = lane >> 4;
  int t   = wave >> 12;       // 0=Q 1=K 2=V
  int rem = wave & 4095;
  int h   = rem >> 9;         // head
  int rb  = rem & 511;        // 16-row block

  const float* X = (t == 0) ? xq : (t == 1) ? xk : xv;
  const float* W = (t == 0) ? Wq : (t == 1) ? Wk : Wv;

  const float* xrow = X + (size_t)(rb * 16 + m) * 512 + h * HDIM;
  v16h a0 = ldfrag_f32(xrow, 0, half);
  v16h a1 = ldfrag_f32(xrow, 32, half);

  v8f acc[4];
#pragma unroll
  for (int ct = 0; ct < 4; ++ct) {
    // B = W^T : B-fragment lane holds row (ct*16+m) of row-major W.
    const float* wrow = W + (size_t)(ct * 16 + m) * HDIM;
    v16h b0 = ldfrag_f32(wrow, 0, half);
    v16h b1 = ldfrag_f32(wrow, 32, half);
    v8f c;
#pragma unroll
    for (int r = 0; r < 8; ++r) c[r] = 0.f;
    c = wmma_f16(a0, b0, c);
    c = wmma_f16(a1, b1, c);
    acc[ct] = c;
  }

  // Scatter from C-layout: VGPR r -> row M = r + 8*half, column N = m.
#pragma unroll
  for (int ct = 0; ct < 4; ++ct) {
#pragma unroll
    for (int r = 0; r < 8; ++r) {
      int row = rb * 16 + r + 8 * half;            // global (n,s) row
      int n = row >> 12, s = row & 4095;
      int d = ct * 16 + m;
      _Float16 val = (_Float16)acc[ct][r];
      if (t == 2) {
        vt[(((size_t)n * HEADS + h) * HDIM + d) * SEQ + s] = val;
      } else {
        _Float16* dst = (t == 0) ? qh : kh;
        dst[(((size_t)n * HEADS + h) * SEQ + s) * HDIM + d] = val;
      }
    }
  }
}

// ---------------------------------------------------------------------------
// Kernel 2: flash attention. 256 threads = 8 waves; each wave owns 16 Q rows.
// K/V tiles stream into double-buffered LDS via gfx1250 async-to-LDS copies
// (ASYNCcnt pipeline), overlapping the next tile's DMA with current WMMAs.
// ---------------------------------------------------------------------------
__global__ __launch_bounds__(256) void attn_kernel(
    const _Float16* __restrict__ qh, const _Float16* __restrict__ kh,
    const _Float16* __restrict__ vt, _Float16* __restrict__ ctx) {
  __shared__ _Float16 kT[2][64 * 64];     // K tile,  row-major [64 rows][64 d]
  __shared__ _Float16 vT[2][64 * 64];     // V^T tile, [64 d][64 k]
  __shared__ _Float16 pB[8][16 * 64];     // per-wave P transpose buffer

  int tid = threadIdx.x, lane = tid & 31, wave = tid >> 5;
  int m = lane & 15, half = lane >> 4;
  int nh = blockIdx.x >> 5;               // (n*8+h)
  int qblk = blockIdx.x & 31;

  const _Float16* Q = qh + (size_t)nh * SEQ * HDIM;
  const _Float16* K = kh + (size_t)nh * SEQ * HDIM;
  const _Float16* V = vt + (size_t)nh * HDIM * SEQ;
  int qbase = qblk * 128 + wave * 16;

  const _Float16* qrow = Q + (size_t)(qbase + m) * HDIM;
  v16h qa0 = ldfrag(qrow, 0, half);
  v16h qa1 = ldfrag(qrow, 32, half);

  v8f o[4];
  float mr[8], lr[8];
#pragma unroll
  for (int dt = 0; dt < 4; ++dt)
#pragma unroll
    for (int r = 0; r < 8; ++r) o[dt][r] = 0.f;
#pragma unroll
  for (int r = 0; r < 8; ++r) { mr[r] = -1.0e30f; lr[r] = 0.f; }

  // Issue async global->LDS copies for tile `tile` into buffer `buf`.
  // 4 wave-instructions per wave (2 for K, 2 for V^T) -> ASYNCcnt += 4.
  auto issue_stage = [&](int tile, int buf) {
    const void* ksrc = (const void*)(K + (size_t)tile * 64 * HDIM);  // 8KB contiguous
    const void* vsrc = (const void*)(V + tile * 64);                 // rows d, stride SEQ
    uint32_t kldsBase = (uint32_t)(uintptr_t)(&kT[buf][0]);
    uint32_t vldsBase = (uint32_t)(uintptr_t)(&vT[buf][0]);
#pragma unroll
    for (int rep = 0; rep < 2; ++rep) {
      uint32_t u = (uint32_t)tid + rep * 256;          // 0..511, one uint4 each
      uint32_t kvoff = u * 16;                         // byte offset in K tile
      uint32_t klds  = kldsBase + u * 16;
      asm volatile("global_load_async_to_lds_b128 %0, %1, %2"
                   :: "v"(klds), "v"(kvoff), "s"(ksrc) : "memory");
      uint32_t d = u >> 3, ko = (u & 7) * 8;           // V^T: [d][k] halves
      uint32_t vvoff = (d * (uint32_t)SEQ + ko) * 2;   // bytes
      uint32_t vlds  = vldsBase + u * 16;
      asm volatile("global_load_async_to_lds_b128 %0, %1, %2"
                   :: "v"(vlds), "v"(vvoff), "s"(vsrc) : "memory");
    }
  };

  issue_stage(0, 0);

  const int NT = SEQ / 64;  // 64 tiles
  for (int it = 0; it < NT; ++it) {
    int buf = it & 1;
    if (it + 1 < NT) {
      issue_stage(it + 1, buf ^ 1);                    // overlap DMA with compute
      asm volatile("s_wait_asynccnt 0x4" ::: "memory"); // tile `it` arrived (ours)
    } else {
      asm volatile("s_wait_asynccnt 0x0" ::: "memory");
    }
    if (it + 2 < NT)  // warm L2 for the tile after next
      __builtin_prefetch((const void*)(K + (size_t)(it + 2) * 64 * HDIM + tid * 16), 0, 0);
    __syncthreads();                                   // all waves' tile arrived

    // ---- S = Q K^T for 4 column tiles (B-frag of K^T == A-frag of K rows)
    v8f st[4];
#pragma unroll
    for (int jt = 0; jt < 4; ++jt) {
      const _Float16* krow = &kT[buf][(jt * 16 + m) * HDIM];
      v16h b0 = ldfrag(krow, 0, half);
      v16h b1 = ldfrag(krow, 32, half);
      v8f c;
#pragma unroll
      for (int r = 0; r < 8; ++r) c[r] = 0.f;
      c = wmma_f16(qa0, b0, c);
      c = wmma_f16(qa1, b1, c);
      st[jt] = c;
    }

    // ---- online softmax (row stats live per VGPR slot; reduce over 16 lanes)
    float tmax[8];
#pragma unroll
    for (int r = 0; r < 8; ++r) {
      float v = st[0][r] * SM_SCALE;
#pragma unroll
      for (int jt = 1; jt < 4; ++jt) v = fmaxf(v, st[jt][r] * SM_SCALE);
      tmax[r] = v;
    }
#pragma unroll
    for (int off = 8; off >= 1; off >>= 1)
#pragma unroll
      for (int r = 0; r < 8; ++r)
        tmax[r] = fmaxf(tmax[r], __shfl_xor(tmax[r], off, 32));

    float alpha[8], rsum[8];
#pragma unroll
    for (int r = 0; r < 8; ++r) {
      float mn = fmaxf(mr[r], tmax[r]);
      alpha[r] = __expf(mr[r] - mn);
      mr[r] = mn;
      rsum[r] = 0.f;
    }
#pragma unroll
    for (int jt = 0; jt < 4; ++jt)
#pragma unroll
      for (int r = 0; r < 8; ++r) {
        float p = __expf(st[jt][r] * SM_SCALE - mr[r]);
        st[jt][r] = p;
        rsum[r] += p;
      }
#pragma unroll
    for (int off = 8; off >= 1; off >>= 1)
#pragma unroll
      for (int r = 0; r < 8; ++r) rsum[r] += __shfl_xor(rsum[r], off, 32);
#pragma unroll
    for (int r = 0; r < 8; ++r) lr[r] = lr[r] * alpha[r] + rsum[r];
#pragma unroll
    for (int dt = 0; dt < 4; ++dt)
#pragma unroll
      for (int r = 0; r < 8; ++r) o[dt][r] *= alpha[r];

    // ---- transpose P (C-layout -> A-frag) through per-wave LDS buffer
    _Float16* pb = pB[wave];
#pragma unroll
    for (int jt = 0; jt < 4; ++jt)
#pragma unroll
      for (int r = 0; r < 8; ++r)
        pb[(r + 8 * half) * 64 + jt * 16 + m] = (_Float16)st[jt][r];
    asm volatile("s_wait_dscnt 0" ::: "memory");  // wave-internal LDS RAW
    v16h pa0 = ldfrag(pb + m * 64, 0, half);
    v16h pa1 = ldfrag(pb + m * 64, 32, half);

    // ---- O += P V (B-frag of V[32k x 16d] == rows of V^T tile)
#pragma unroll
    for (int dt = 0; dt < 4; ++dt) {
      const _Float16* vrow = &vT[buf][(dt * 16 + m) * 64];
      v16h b0 = ldfrag(vrow, 0, half);
      v16h b1 = ldfrag(vrow, 32, half);
      o[dt] = wmma_f16(pa0, b0, o[dt]);
      o[dt] = wmma_f16(pa1, b1, o[dt]);
    }
    __syncthreads();  // WAR: next iteration's async copies overwrite buf^1... / buf
  }

  // ---- epilogue: normalize, store f16 context [N,S,H*64]
  int n = nh >> 3, h = nh & 7;
#pragma unroll
  for (int dt = 0; dt < 4; ++dt)
#pragma unroll
    for (int r = 0; r < 8; ++r) {
      int s = qbase + r + 8 * half;
      int d = dt * 16 + m;
      float val = o[dt][r] / lr[r];
      ctx[(size_t)(n * SEQ + s) * 512 + h * HDIM + d] = (_Float16)val;
    }
}

// ---------------------------------------------------------------------------
// Kernel 3: out = ctx @ Wfc^T + bfc  (M=8192, N=512, K=512), f32 out.
// One wave: 16 rows x 64 cols.
// ---------------------------------------------------------------------------
__global__ __launch_bounds__(256) void fc_kernel(
    const _Float16* __restrict__ ctx, const _Float16* __restrict__ wfch,
    const float* __restrict__ bfc, float* __restrict__ out) {
  int wave = blockIdx.x * 8 + (threadIdx.x >> 5);   // 0..4095
  int lane = threadIdx.x & 31;
  int m = lane & 15, half = lane >> 4;
  int rt = wave >> 3;     // 16-row tile
  int cg = wave & 7;      // 64-col group

  const _Float16* arow = ctx + (size_t)(rt * 16 + m) * 512;
  v8f acc[4];
#pragma unroll
  for (int ct = 0; ct < 4; ++ct)
#pragma unroll
    for (int r = 0; r < 8; ++r) acc[ct][r] = 0.f;

#pragma unroll 4
  for (int kk = 0; kk < 16; ++kk) {
    int kBase = kk * 32;
    v16h a = ldfrag(arow, kBase, half);
#pragma unroll
    for (int ct = 0; ct < 4; ++ct) {
      const _Float16* wrow = wfch + (size_t)(cg * 64 + ct * 16 + m) * 512;
      v16h b = ldfrag(wrow, kBase, half);
      acc[ct] = wmma_f16(a, b, acc[ct]);
    }
  }
#pragma unroll
  for (int ct = 0; ct < 4; ++ct)
#pragma unroll
    for (int r = 0; r < 8; ++r) {
      int row = rt * 16 + r + 8 * half;
      int col = cg * 64 + ct * 16 + m;
      out[(size_t)row * 512 + col] = acc[ct][r] + bfc[col];
    }
}

// ---------------------------------------------------------------------------
extern "C" void kernel_launch(void* const* d_in, const int* in_sizes, int n_in,
                              void* d_out, int out_size, void* d_ws, size_t ws_size,
                              hipStream_t stream) {
  const float* x_q = (const float*)d_in[0];
  const float* x_k = (const float*)d_in[1];
  const float* x_v = (const float*)d_in[2];
  const float* Wq  = (const float*)d_in[3];
  const float* Wk  = (const float*)d_in[4];
  const float* Wv  = (const float*)d_in[5];
  const float* Wfc = (const float*)d_in[6];
  const float* bfc = (const float*)d_in[7];
  float* out = (float*)d_out;

  const size_t TSZ = (size_t)NBATCH * HEADS * SEQ * HDIM * sizeof(_Float16); // 8 MB
  char* ws = (char*)d_ws;
  _Float16* qh   = (_Float16*)(ws);
  _Float16* kh   = (_Float16*)(ws + TSZ);
  _Float16* vt   = (_Float16*)(ws + 2 * TSZ);
  _Float16* ctx  = (_Float16*)(ws + 3 * TSZ);
  _Float16* wfch = (_Float16*)(ws + 4 * TSZ);

  cvt_kernel<<<(512 * 512) / 256, 256, 0, stream>>>(Wfc, wfch);
  proj_kernel<<<1536, 256, 0, stream>>>(x_q, x_k, x_v, Wq, Wk, Wv, qh, kh, vt);
  attn_kernel<<<512, 256, 0, stream>>>(qh, kh, vt, ctx);
  fc_kernel<<<512, 256, 0, stream>>>(ctx, wfch, bfc, out);
}
